// EMAVectorQuantizer_10230612099576
// MI455X (gfx1250) — compile-verified, matrix operand order checked by reading, not run on hardware
//
#include <hip/hip_runtime.h>
#include <hip/hip_bf16.h>
#include <math.h>

// ---------------- problem constants ----------------
#define DECAY_ 0.99f
#define ALPHA_ 0.01f   /* 1 - DECAY */
#define EPS_   1e-5f
#define BETA_  0.25f

static constexpr int N_TOK   = 8192;     // b*h*w
static constexpr int DIM     = 256;
static constexpr int K_CODES = 16384;
static constexpr int HW      = 1024;     // h*w

// output offsets (in floats), concatenated in reference return order
static constexpr long long OFF_ZQ   = 0;          // (8,256,32,32)
static constexpr long long OFF_LOSS = 2097152;    // scalar
static constexpr long long OFF_PERP = 2097153;    // scalar
static constexpr long long OFF_IDX  = 2097154;    // (8,32,32)
static constexpr long long OFF_W    = 2105346;    // (16384,256)
static constexpr long long OFF_CS   = 6299650;    // (16384,)
static constexpr long long OFF_EA   = 6316034;    // (16384,256)

typedef __attribute__((ext_vector_type(16))) __bf16        v16bf;
typedef __attribute__((ext_vector_type(8)))  float         v8f;
typedef __attribute__((ext_vector_type(4)))  unsigned int  v4u;
typedef __attribute__((ext_vector_type(8)))  int           v8i;
typedef __attribute__((ext_vector_type(4)))  int           v4i;

// ---------------- TDM helper ----------------
// Tensor DMA: load a (32 codes x 256 dims) bf16 tile (16 KB) from the bf16
// codebook into LDS. D# layout per CDNA5 ISA ch.8:
//   group0: count=1 | lds_addr | global_addr[56:0] | type=2
//   group1: data_size=1(2B), tensor_dim0=256, tensor_dim1=16384,
//           tile_dim0=256, tile_dim1=32, tensor_dim0_stride=256
__device__ __forceinline__ void tdm_load_tile(unsigned lds_off, const __bf16* gptr) {
    unsigned long long ga = (unsigned long long)(uintptr_t)gptr;
    v4u g0 = { 1u,                                  // count=1, user desc
               lds_off,                             // LDS byte address
               (unsigned)ga,                        // global_addr[31:0]
               (unsigned)(ga >> 32) | 0x80000000u };// global_addr[56:32] | type=2
    v8i g1 = { 0x00010000,            // wg_mask=0, data_size=1 (2 bytes)
               0x01000000,            // tensor_dim0 = 256 (bits 79:48, low part)
               0x40000000,            // tensor_dim1 = 16384 (bits 111:80, low part)
               0x01000000,            // tile_dim0 = 256 (bits 127:112)
               32,                    // tile_dim1 = 32 rows
               256,                   // tensor_dim0_stride = 256 (bits 207:160)
               0, 0 };
    v4i g2 = { 0, 0, 0, 0 };
    v4i g3 = { 0, 0, 0, 0 };
#if __clang_major__ >= 23
    v8i g4 = { 0, 0, 0, 0, 0, 0, 0, 0 };
    __builtin_amdgcn_tensor_load_to_lds(g0, g1, g2, g3, g4, 0);
#else
    __builtin_amdgcn_tensor_load_to_lds(g0, g1, g2, g3, 0);
#endif
}

// ---------------- kernels ----------------

// init: new_cluster_size = DECAY*cs, zero counts/scalars
__global__ __launch_bounds__(256) void vq_init_k(const float* __restrict__ cs,
                                                 float* __restrict__ out_cs,
                                                 float* __restrict__ counts,
                                                 float* __restrict__ scal) {
    int k = blockIdx.x * 256 + threadIdx.x;
    out_cs[k] = DECAY_ * cs[k];
    counts[k] = 0.0f;
    if (k < 4) scal[k] = 0.0f;
}

// per-code ||e||^2 and weight -> bf16 (row-major)
__global__ __launch_bounds__(256) void vq_prep_w(const float* __restrict__ w,
                                                 __bf16* __restrict__ wbf,
                                                 float* __restrict__ enorm) {
    int lane = threadIdx.x & 31, wave = threadIdx.x >> 5;
    int code = blockIdx.x * 8 + wave;
    const float* row = w + (long long)code * DIM;
    float4 f0 = ((const float4*)row)[lane * 2 + 0];
    float4 f1 = ((const float4*)row)[lane * 2 + 1];
    alignas(16) __bf16 hb[8];
    hb[0] = (__bf16)f0.x; hb[1] = (__bf16)f0.y; hb[2] = (__bf16)f0.z; hb[3] = (__bf16)f0.w;
    hb[4] = (__bf16)f1.x; hb[5] = (__bf16)f1.y; hb[6] = (__bf16)f1.z; hb[7] = (__bf16)f1.w;
    *(uint4*)(wbf + (long long)code * DIM + lane * 8) = *(const uint4*)hb;
    float s = f0.x*f0.x + f0.y*f0.y + f0.z*f0.z + f0.w*f0.w
            + f1.x*f1.x + f1.y*f1.y + f1.z*f1.z + f1.w*f1.w;
    #pragma unroll
    for (int m = 16; m >= 1; m >>= 1) s += __shfl_xor(s, m, 32);
    if (lane == 0) enorm[code] = s;
}

// z (b,c,h,w) -> zf bf16 (n, d) with n = b*1024 + h*32 + w
__global__ __launch_bounds__(256) void vq_prep_z(const float* __restrict__ z,
                                                 __bf16* __restrict__ zbf) {
    int n = blockIdx.x, d = threadIdx.x;
    int b = n >> 10, hw = n & 1023;
    float v = z[((long long)b * DIM + d) * HW + hw];
    zbf[(long long)n * DIM + d] = (__bf16)v;
}

// WMMA distance + argmin. 64 blocks x 256 threads (8 waves), 16 rows/wave.
// Weight tiles (32 codes x 256 dims bf16 = 16KB) streamed into a double-
// buffered LDS region by the Tensor Data Mover, overlapped with WMMA compute.
__global__ __launch_bounds__(256) void vq_argmin(const __bf16* __restrict__ zbf,
                                                 const __bf16* __restrict__ wbf,
                                                 const float* __restrict__ enorm,
                                                 int* __restrict__ idx_out,
                                                 float* __restrict__ idxf_out) {
    __shared__ __bf16 tiles[2][32 * DIM];   // 2 x 16KB double buffer
    const int tid  = threadIdx.x;
    const int lane = tid & 31;
    const int wave = tid >> 5;
    const int half = lane >> 4;             // 0: lanes 0-15, 1: lanes 16-31
    const int col  = lane & 15;
    const int rowBase = blockIdx.x * 128 + wave * 16;

    const unsigned lds_base = (unsigned)(uintptr_t)(void*)&tiles[0][0];

    // A fragments: 16 token rows x 256 dims, resident for whole K loop
    v16bf a[8];
    {
        const __bf16* zr = zbf + (long long)(rowBase + col) * DIM + half * 16;
        #pragma unroll
        for (int c = 0; c < 8; ++c) a[c] = *(const v16bf*)(zr + c * 32);
    }

    float best[8]; int bidx[8];
    #pragma unroll
    for (int j = 0; j < 8; ++j) { best[j] = 3.4e38f; bidx[j] = 0; }

    const int NT = K_CODES / 32;            // 512 k-tiles of 32 codes
    if (wave == 0) tdm_load_tile(lds_base, wbf);   // prologue: tile 0 -> buf 0

    for (int kt = 0; kt < NT; ++kt) {
        if (wave == 0) {
            if (kt + 1 < NT) {
                tdm_load_tile(lds_base + (unsigned)(((kt + 1) & 1) * 16384),
                              wbf + (long long)(kt + 1) * 32 * DIM);
                __builtin_amdgcn_s_wait_tensorcnt(1);  // tile kt landed
            } else {
                __builtin_amdgcn_s_wait_tensorcnt(0);
            }
        }
        __syncthreads();                     // tile kt visible to all waves

        const __bf16* tbuf = &tiles[kt & 1][0];
        v8f acc0 = {}, acc1 = {};
        #pragma unroll
        for (int c = 0; c < 8; ++c) {
            v16bf b0 = *(const v16bf*)(tbuf + col * DIM        + c * 32 + half * 16);
            v16bf b1 = *(const v16bf*)(tbuf + (16 + col) * DIM + c * 32 + half * 16);
            acc0 = __builtin_amdgcn_wmma_f32_16x16x32_bf16(
                false, a[c], false, b0, (short)0, acc0, false, false);
            acc1 = __builtin_amdgcn_wmma_f32_16x16x32_bf16(
                false, a[c], false, b1, (short)0, acc1, false, false);
        }

        int   code0 = kt * 32 + col;
        int   code1 = code0 + 16;
        float en0   = enorm[code0];
        float en1   = enorm[code1];
        #pragma unroll
        for (int j = 0; j < 8; ++j) {
            float s0 = en0 - 2.0f * acc0[j];   // ||e||^2 - 2 z.e
            float s1 = en1 - 2.0f * acc1[j];
            if (s0 < best[j]) { best[j] = s0; bidx[j] = code0; }
            if (s1 < best[j]) { best[j] = s1; bidx[j] = code1; }
        }
        __syncthreads();                     // reads done before buffer reuse
    }

    // cross-lane argmin within each 16-lane group (C layout: lane = code column)
    #pragma unroll
    for (int j = 0; j < 8; ++j) {
        float b = best[j]; int i = bidx[j];
        #pragma unroll
        for (int m = 8; m >= 1; m >>= 1) {
            float ob = __shfl_xor(b, m, 32);
            int   oi = __shfl_xor(i, m, 32);
            if (ob < b || (ob == b && oi < i)) { b = ob; i = oi; }
        }
        if (col == 0) {
            int row = rowBase + half * 8 + j;
            idx_out[row]  = i;
            idxf_out[row] = (float)i;
        }
    }
}

// new_embed_avg base = DECAY * embed_avg
__global__ __launch_bounds__(256) void vq_init_ea(const float* __restrict__ ea,
                                                  float* __restrict__ out_ea) {
    long long i = (long long)blockIdx.x * 256 + threadIdx.x;
    out_ea[i] = DECAY_ * ea[i];
}

// scatter EMA: out_ea[k,:] += (1-DECAY)*z_n ; out_cs[k] += (1-DECAY); counts[k]++
__global__ __launch_bounds__(256) void vq_scatter(const float* __restrict__ z,
                                                  const int* __restrict__ idx,
                                                  float* __restrict__ out_ea,
                                                  float* __restrict__ out_cs,
                                                  float* __restrict__ counts) {
    int n = blockIdx.x, d = threadIdx.x;
    int k = idx[n];
    int b = n >> 10, hw = n & 1023;
    float v = z[((long long)b * DIM + d) * HW + hw];
    atomicAdd(&out_ea[(long long)k * DIM + d], ALPHA_ * v);
    if (d == 0) {
        atomicAdd(&out_cs[k], ALPHA_);
        atomicAdd(&counts[k], 1.0f);
    }
}

// z_q gather (straight-through output == z_q) + commitment-loss accumulation
__global__ __launch_bounds__(256) void vq_zq_loss(const float* __restrict__ z,
                                                  const float* __restrict__ w,
                                                  const int* __restrict__ idx,
                                                  float* __restrict__ out_zq,
                                                  float* __restrict__ scal) {
    __shared__ float red[256];
    int n = blockIdx.x, d = threadIdx.x;
    int k = idx[n];
    int b = n >> 10, hw = n & 1023;
    long long zi = ((long long)b * DIM + d) * HW + hw;
    float wv = w[(long long)k * DIM + d];
    out_zq[zi] = wv;
    float diff = wv - z[zi];
    red[d] = diff * diff;
    __syncthreads();
    #pragma unroll
    for (int s = 128; s > 0; s >>= 1) {
        if (d < s) red[d] += red[d + s];
        __syncthreads();
    }
    if (d == 0) atomicAdd(&scal[0], red[0]);
}

// scalar stats: n_total, perplexity, final loss
__global__ __launch_bounds__(256) void vq_stats(const float* __restrict__ out_cs,
                                                const float* __restrict__ counts,
                                                float* __restrict__ scal,
                                                float* __restrict__ out_loss,
                                                float* __restrict__ out_perp) {
    __shared__ float rnt[256], rent[256];
    int t = threadIdx.x;
    float nt = 0.0f, ent = 0.0f;
    for (int k = t; k < K_CODES; k += 256) {
        nt += out_cs[k];
        float p = counts[k] * (1.0f / (float)N_TOK);
        ent += p * logf(p + 1e-10f);
    }
    rnt[t] = nt; rent[t] = ent;
    __syncthreads();
    #pragma unroll
    for (int s = 128; s > 0; s >>= 1) {
        if (t < s) { rnt[t] += rnt[t + s]; rent[t] += rent[t + s]; }
        __syncthreads();
    }
    if (t == 0) {
        scal[1] = rnt[0];
        *out_loss = BETA_ * scal[0] * (1.0f / 2097152.0f);
        *out_perp = expf(-rent[0]);
    }
}

// new_weight = new_embed_avg / smoothed
__global__ __launch_bounds__(256) void vq_new_w(const float* __restrict__ out_ea,
                                                const float* __restrict__ out_cs,
                                                const float* __restrict__ scal,
                                                float* __restrict__ out_w) {
    long long i = (long long)blockIdx.x * 256 + threadIdx.x;
    int k = (int)(i >> 8);
    float nt = scal[1];
    float sm = (out_cs[k] + EPS_) / (nt + (float)K_CODES * EPS_) * nt;
    out_w[i] = out_ea[i] / sm;
}

// ---------------- launch ----------------
extern "C" void kernel_launch(void* const* d_in, const int* in_sizes, int n_in,
                              void* d_out, int out_size, void* d_ws, size_t ws_size,
                              hipStream_t stream) {
    const float* z  = (const float*)d_in[0];   // (8,256,32,32)
    const float* w  = (const float*)d_in[1];   // (16384,256)
    const float* cs = (const float*)d_in[2];   // (16384,)
    const float* ea = (const float*)d_in[3];   // (16384,256)
    float* out = (float*)d_out;

    char* ws = (char*)d_ws;
    __bf16* zbf   = (__bf16*)(ws);                        // 4 MB
    __bf16* wbf   = (__bf16*)(ws + 4194304);              // 8 MB
    float*  enorm = (float*)(ws + 12582912);              // 64 KB
    int*    idx   = (int*)  (ws + 12648448);              // 32 KB
    float*  counts= (float*)(ws + 12681216);              // 64 KB
    float*  scal  = (float*)(ws + 12746752);              // scalars

    vq_init_k <<<K_CODES / 256, 256, 0, stream>>>(cs, out + OFF_CS, counts, scal);
    vq_prep_w <<<K_CODES / 8,  256, 0, stream>>>(w, wbf, enorm);
    vq_prep_z <<<N_TOK,        256, 0, stream>>>(z, zbf);
    vq_argmin <<<N_TOK / 128,  256, 0, stream>>>(zbf, wbf, enorm, idx, out + OFF_IDX);
    vq_init_ea<<<K_CODES,      256, 0, stream>>>(ea, out + OFF_EA);
    vq_scatter<<<N_TOK,        256, 0, stream>>>(z, idx, out + OFF_EA, out + OFF_CS, counts);
    vq_zq_loss<<<N_TOK,        256, 0, stream>>>(z, w, idx, out + OFF_ZQ, scal);
    vq_stats  <<<1,            256, 0, stream>>>(out + OFF_CS, counts, scal,
                                                 out + OFF_LOSS, out + OFF_PERP);
    vq_new_w  <<<K_CODES,      256, 0, stream>>>(out + OFF_EA, out + OFF_CS, scal, out + OFF_W);
}
